// RNNBase_46188078301330
// MI455X (gfx1250) — compile-verified, matrix operand order checked by reading, not run on hardware
//
#include <hip/hip_runtime.h>
#include <hip/hip_bf16.h>
#include <math.h>
#include <stdint.h>

typedef __bf16 bf16_t;
typedef bf16_t v16bf __attribute__((ext_vector_type(16)));
typedef bf16_t v8bf  __attribute__((ext_vector_type(8)));
typedef float  v8f   __attribute__((ext_vector_type(8)));

static constexpr int BB  = 64;           // batch
static constexpr int T   = 512;          // timesteps
static constexpr int D   = 512;          // input dim
static constexpr int H   = 1024;         // hidden
static constexpr int O   = 512;          // output dim
static constexpr int G3H = 3 * H;        // 3072
static constexpr long MT = (long)BB * T; // 32768 rows for the big GEMMs

// ---------------------------------------------------------------------------
// 16x32 bf16 WMMA fragment (A-style striping, ISA 7.12.2):
//   lane l holds row (row0 + l%16); lane-half g = l/16 selects K sub-bands:
//   elements 0..7  -> K = k0 + 8*g + (0..7)
//   elements 8..15 -> K = k0 + 16 + 8*g + (0..7)
// Two 16B loads per fragment; global_load_b128 or (after addrspace
// inference) ds_load_b128 for LDS-staged tiles.
// W matrices are stored (N,K) row-major, i.e. already transposed for B.
// ---------------------------------------------------------------------------
__device__ __forceinline__ v16bf load_frag(const bf16_t* __restrict__ p,
                                           int ld, int row, int k0, int half) {
  const bf16_t* q = p + (size_t)row * ld + k0 + half * 8;
  union { v16bf v; v8bf h[2]; } f;
  f.h[0] = *(const v8bf*)(q);
  f.h[1] = *(const v8bf*)(q + 16);
  return f.v;
}

__device__ __forceinline__ v8f wmma_bf16(v16bf a, v16bf b, v8f c) {
  return __builtin_amdgcn_wmma_f32_16x16x32_bf16(false, a, false, b, (short)0, c,
                                                 false, false);
}

__device__ __forceinline__ float sigmoidf_dev(float x) {
  return 1.0f / (1.0f + __expf(-x));
}

// ---------------------------------------------------------------------------
// fp32 -> bf16 streaming conversion (8 elements / thread).
// ---------------------------------------------------------------------------
__global__ void cvt_f32_bf16_kernel(const float* __restrict__ s,
                                    bf16_t* __restrict__ d, long n) {
  long i = ((long)blockIdx.x * blockDim.x + threadIdx.x) * 8;
  if (i >= n) return;
  float4 a = *(const float4*)(s + i);
  float4 b = *(const float4*)(s + i + 4);
  v8bf o;
  o[0] = (bf16_t)a.x; o[1] = (bf16_t)a.y; o[2] = (bf16_t)a.z; o[3] = (bf16_t)a.w;
  o[4] = (bf16_t)b.x; o[5] = (bf16_t)b.y; o[6] = (bf16_t)b.z; o[7] = (bf16_t)b.w;
  *(v8bf*)(d + i) = o;
}

__global__ void zero_bf16_kernel(bf16_t* __restrict__ p, int n) {
  int i = blockIdx.x * blockDim.x + threadIdx.x;
  if (i < n) p[i] = (bf16_t)0.0f;
}

__global__ void zero_u32_kernel(unsigned* __restrict__ p, int n) {
  int i = blockIdx.x * blockDim.x + threadIdx.x;
  if (i < n) p[i] = 0u;
}

// ---------------------------------------------------------------------------
// GEMM: C[M,N] = A[M,K] * W[N,K]^T + bias[N].  bf16 in, fp32 accumulate.
// Block = 256 threads = 8 waves; a block owns ONE 16-row A tile (staged in
// LDS once, shared by all 8 waves) and 8 consecutive 64-wide N strips; each
// wave computes a 16x64 strip: 1 LDS A-fragment feeds 4 WMMAs per K-slab.
// Grid: (M/16) * (N/64/8) blocks.  OUT_BF16 selects bf16 or fp32 output.
// ---------------------------------------------------------------------------
template <int N_, int K_, bool OUT_BF16>
__global__ void gemm_bias_kernel(const bf16_t* __restrict__ A,
                                 const bf16_t* __restrict__ W,
                                 const float* __restrict__ bias,
                                 void* __restrict__ Cout) {
  __shared__ __align__(16) bf16_t sA[16 * K_];

  const int lane = threadIdx.x & 31;
  const int wave = threadIdx.x >> 5;
  const int half = lane >> 4;
  const int lrow = lane & 15;

  constexpr int stripsN    = N_ / 64;
  constexpr int blocksPerM = stripsN / 8;
  const int tm = blockIdx.x / blocksPerM;
  const int bn = blockIdx.x % blocksPerM;
  const int tn4 = bn * 8 + wave;

  {
    const uint4* src = (const uint4*)(A + (size_t)tm * 16 * K_);
    uint4* dst = (uint4*)sA;
    constexpr int nvec = 16 * K_ / 8;
#pragma unroll
    for (int i = 0; i < nvec / 256; ++i)
      dst[i * 256 + threadIdx.x] = src[i * 256 + threadIdx.x];
  }
  __syncthreads();

  const int rowB0 = tn4 * 64 + lrow;

  v8f acc0 = {}, acc1 = {}, acc2 = {}, acc3 = {};
  for (int k0 = 0; k0 < K_; k0 += 32) {
    v16bf af = load_frag(sA, K_, lrow, k0, half);   // LDS
    v16bf b0 = load_frag(W, K_, rowB0,      k0, half);
    acc0 = wmma_bf16(af, b0, acc0);
    v16bf b1 = load_frag(W, K_, rowB0 + 16, k0, half);
    acc1 = wmma_bf16(af, b1, acc1);
    v16bf b2 = load_frag(W, K_, rowB0 + 32, k0, half);
    acc2 = wmma_bf16(af, b2, acc2);
    v16bf b3 = load_frag(W, K_, rowB0 + 48, k0, half);
    acc3 = wmma_bf16(af, b3, acc3);
  }

  v8f* accs[4] = {&acc0, &acc1, &acc2, &acc3};
#pragma unroll
  for (int j = 0; j < 4; ++j) {
    int n = tn4 * 64 + j * 16 + lrow;
    float bn_ = bias[n];
#pragma unroll
    for (int r = 0; r < 8; ++r) {
      int m = tm * 16 + r + 8 * half;
      float v = (*accs[j])[r] + bn_;
      if (OUT_BF16)
        ((bf16_t*)Cout)[(size_t)m * N_ + n] = (bf16_t)v;
      else
        ((float*)Cout)[(size_t)m * N_ + n] = v;
    }
  }
}

// ---------------------------------------------------------------------------
// Persistent GRU recurrence: ONE launch for all 512 timesteps.
// 32 co-resident blocks loop over t; each step:
//   - restage the block's 16-row h_prev tile (16x1024 bf16 = 32 KB) in LDS
//   - 8 waves x (r/u/n 16x16 tiles), 3 WMMAs per 32-K slab off 1 LDS A frag
//   - fused sigmoid/tanh gates; h ping-pong write; hiddens fp32 + bf16 out
//   - device-scope sense-reversing grid barrier (replaces a kernel launch)
// A laundered ZERO OFFSET (not the pointer) defeats LICM of the Wh fragment
// loads (which previously spilled 3 KB/lane to scratch) while preserving the
// pointer's address-space provenance so they stay global_load_b128.
// ---------------------------------------------------------------------------
__global__ void gru_persistent_kernel(const bf16_t* __restrict__ Wh,     // [3H,H]
                                      const float*  __restrict__ bh,     // [3H]
                                      const bf16_t* __restrict__ x_all,  // [BB,T,3H]
                                      bf16_t* __restrict__ h0,           // [BB,H]
                                      bf16_t* __restrict__ h1,           // [BB,H]
                                      float*  __restrict__ hiddens,      // [BB,T,H]
                                      bf16_t* __restrict__ hid_bf,       // [BB,T,H]
                                      unsigned* __restrict__ bar) {      // [64]
  __shared__ __align__(16) bf16_t sH[16 * H];

  const int lane = threadIdx.x & 31;
  const int wave = threadIdx.x >> 5;
  const int half = lane >> 4;
  const int lrow = lane & 15;

  const int id = blockIdx.x * 8 + wave;   // 0..255
  const int tj = id & 63;
  const int tm = id >> 6;                 // constant within a block (8 | 64)

  const int rowR = tj * 16 + lrow;
  const int rowU = H + rowR;
  const int rowN = 2 * H + rowR;
  const int j = tj * 16 + lrow;
  const float bhr = bh[j];
  const float bhu = bh[H + j];
  const float bhn = bh[2 * H + j];

  unsigned* counter = bar;                // cacheline 0
  unsigned* flag    = bar + 32;           // cacheline 1

  for (int t = 0; t < T; ++t) {
    const bf16_t* hp = (t & 1) ? h1 : h0;
    bf16_t*       hn = (t & 1) ? h0 : h1;

    // Launder a zero offset: Wh loads not provably loop-invariant (no LICM
    // mega-hoist/spill), but pointer provenance intact -> global_load.
    size_t zoff = 0;
    asm volatile("" : "+s"(zoff));
    const bf16_t* wh_t = Wh + zoff;

    // Stage h_prev rows [tm*16, tm*16+16) into LDS.
    {
      const uint4* src = (const uint4*)(hp + (size_t)tm * 16 * H);
      uint4* dst = (uint4*)sH;
#pragma unroll
      for (int i = 0; i < (16 * H / 8) / 256; ++i)
        dst[i * 256 + threadIdx.x] = src[i * 256 + threadIdx.x];
    }
    __syncthreads();

    v8f cr = {}, cu = {}, cn = {};
    for (int k0 = 0; k0 < H; k0 += 32) {
      v16bf af  = load_frag(sH, H, lrow, k0, half);   // LDS
      v16bf br  = load_frag(wh_t, H, rowR, k0, half);
      cr = wmma_bf16(af, br, cr);
      v16bf bu  = load_frag(wh_t, H, rowU, k0, half);
      cu = wmma_bf16(af, bu, cu);
      v16bf bn_ = load_frag(wh_t, H, rowN, k0, half);
      cn = wmma_bf16(af, bn_, cn);
    }

#pragma unroll
    for (int r = 0; r < 8; ++r) {
      int lrow2 = r + 8 * half;           // local batch row 0..15
      int b = tm * 16 + lrow2;            // batch index 0..63
      const bf16_t* xrow = x_all + ((size_t)b * T + t) * G3H;
      float xr = (float)xrow[j];
      float xu = (float)xrow[H + j];
      float xn = (float)xrow[2 * H + j];
      float hr = cr[r] + bhr;
      float hu = cu[r] + bhu;
      float hh = cn[r] + bhn;
      float reset = sigmoidf_dev(xr + hr);
      float upd   = sigmoidf_dev(xu + hu);
      float nw    = tanhf(xn + reset * hh);
      float hpv   = (float)sH[(size_t)lrow2 * H + j];
      float hy    = upd * hpv + (1.0f - upd) * nw;
      hn[(size_t)b * H + j] = (bf16_t)hy;
      hiddens[((size_t)b * T + t) * H + j] = hy;
      hid_bf[((size_t)b * T + t) * H + j] = (bf16_t)hy;
    }

    // Grid barrier: publish h_next, wait for all 32 blocks.
    __syncthreads();
    if (threadIdx.x == 0) {
      __threadfence();                                   // release h_next
      unsigned arrived = __hip_atomic_fetch_add(counter, 1u, __ATOMIC_ACQ_REL,
                                                __HIP_MEMORY_SCOPE_AGENT);
      if (arrived == gridDim.x - 1) {
        __hip_atomic_store(counter, 0u, __ATOMIC_RELAXED,
                           __HIP_MEMORY_SCOPE_AGENT);    // reset before release
        __hip_atomic_store(flag, (unsigned)(t + 1), __ATOMIC_RELEASE,
                           __HIP_MEMORY_SCOPE_AGENT);
      } else {
        while (__hip_atomic_load(flag, __ATOMIC_ACQUIRE,
                                 __HIP_MEMORY_SCOPE_AGENT) <= (unsigned)t)
          __builtin_amdgcn_s_sleep(1);
      }
      __threadfence();                                   // acquire h_next
    }
    __syncthreads();
  }
}

// ---------------------------------------------------------------------------
extern "C" void kernel_launch(void* const* d_in, const int* in_sizes, int n_in,
                              void* d_out, int out_size, void* d_ws, size_t ws_size,
                              hipStream_t stream) {
  const float* inputs = (const float*)d_in[0];  // [BB,T,D]
  const float* Wx     = (const float*)d_in[1];  // [3H,D]
  const float* bx     = (const float*)d_in[2];  // [3H]
  const float* Wh     = (const float*)d_in[3];  // [3H,H]
  const float* bh     = (const float*)d_in[4];  // [3H]
  const float* Wo     = (const float*)d_in[5];  // [O,H]
  const float* bo     = (const float*)d_in[6];  // [O]

  float* hiddens = (float*)d_out;               // [BB,T,H] fp32
  float* out2    = hiddens + (size_t)MT * H;    // [BB,T,O] fp32

  // Workspace layout (bf16 chunks, 16B-aligned).
  bf16_t* x_all  = (bf16_t*)d_ws;                   // MT*3H
  bf16_t* in_bf  = x_all  + (size_t)MT * G3H;       // MT*D
  bf16_t* wx_bf  = in_bf  + (size_t)MT * D;         // 3H*D
  bf16_t* wh_bf  = wx_bf  + (size_t)G3H * D;        // 3H*H
  bf16_t* wo_bf  = wh_bf  + (size_t)G3H * H;        // O*H
  bf16_t* hid_bf = wo_bf  + (size_t)O * H;          // MT*H
  bf16_t* h0     = hid_bf + (size_t)MT * H;         // BB*H
  bf16_t* h1     = h0 + (size_t)BB * H;             // BB*H
  unsigned* bar  = (unsigned*)(h1 + (size_t)BB * H);// 64 u32 (2 cachelines)

  // One-time fp32 -> bf16 conversions (weights stay L2-resident afterwards).
  {
    long n;
    n = MT * (long)D;
    cvt_f32_bf16_kernel<<<(int)(n / 8 / 256), 256, 0, stream>>>(inputs, in_bf, n);
    n = (long)G3H * D;
    cvt_f32_bf16_kernel<<<(int)(n / 8 / 256), 256, 0, stream>>>(Wx, wx_bf, n);
    n = (long)G3H * H;
    cvt_f32_bf16_kernel<<<(int)(n / 8 / 256), 256, 0, stream>>>(Wh, wh_bf, n);
    n = (long)O * H;
    cvt_f32_bf16_kernel<<<(int)(n / 8 / 256), 256, 0, stream>>>(Wo, wo_bf, n);
  }

  // h(0) = 0; barrier state = 0 (every call -> deterministic).
  zero_bf16_kernel<<<(BB * H + 255) / 256, 256, 0, stream>>>(h0, BB * H);
  zero_u32_kernel<<<1, 64, 0, stream>>>(bar, 64);

  // x_all = inputs @ Wx^T + bx : (32768,512)x(512,3072), bf16 out
  {
    int blocks = (int)(MT / 16) * (G3H / 64 / 8);   // 2048 * 6 = 12288
    gemm_bias_kernel<G3H, D, true>
        <<<blocks, 256, 0, stream>>>(in_bf, wx_bf, bx, x_all);
  }

  // Recurrence: single persistent launch, grid barrier between timesteps.
  gru_persistent_kernel<<<32, 256, 0, stream>>>(wh_bf, bh, x_all, h0, h1,
                                                hiddens, hid_bf, bar);

  // out = hiddens @ Wo^T + bo : (32768,1024)x(1024,512), fp32 out
  {
    int blocks = (int)(MT / 16) * (O / 64 / 8);     // 2048 * 1 = 2048
    gemm_bias_kernel<O, H, false>
        <<<blocks, 256, 0, stream>>>(hid_bf, wo_bf, bo, out2);
  }
}
